// PriorFusion3D_crossattn_17119739642231
// MI455X (gfx1250) — matrix operand order repaired and checked by reading, not compiled, not run on hardware
//
#include <hip/hip_runtime.h>
#include <cstdint>

// ---------------------------------------------------------------------------
// PriorFusion3D cross-attention pipeline for MI455X (gfx1250, wave32, WMMA).
// GEMMs on v_wmma_f32_16x16x32_f16; weight tiles staged to LDS via
// GLOBAL_LOAD_ASYNC_TO_LDS_B128 (ASYNCcnt) when the builtin exists.
// ---------------------------------------------------------------------------

typedef __attribute__((ext_vector_type(16))) _Float16 v16h;
typedef __attribute__((ext_vector_type(8)))  _Float16 v8h;
typedef __attribute__((ext_vector_type(8)))  float    v8f;
typedef __attribute__((ext_vector_type(4)))  int      v4i;

#define WMMA_F16(a, b, c) \
  __builtin_amdgcn_wmma_f32_16x16x32_f16(false, (a), false, (b), (short)0, (c), false, false)

#if defined(__has_builtin)
#if __has_builtin(__builtin_amdgcn_global_load_async_to_lds_b128) && \
    __has_builtin(__builtin_amdgcn_s_wait_asynccnt)
#define ASYNC_LDS 1
#endif
#endif
#ifndef ASYNC_LDS
#define ASYNC_LDS 0
#endif

#if ASYNC_LDS
typedef __attribute__((address_space(1))) v4i* gv4ip;  // pointer to global int4
typedef __attribute__((address_space(3))) v4i* lv4ip;  // pointer to LDS int4
#endif

// copy 16 bytes (8 halves) global -> LDS; async path uses ASYNCcnt
__device__ __forceinline__ void cp16_async(const _Float16* g, _Float16* l) {
#if ASYNC_LDS
  __builtin_amdgcn_global_load_async_to_lds_b128((gv4ip)(g), (lv4ip)(l), 0, 0);
#else
  *(v8h*)l = *(const v8h*)g;
#endif
}
__device__ __forceinline__ void async_fence() {
#if ASYNC_LDS
  __builtin_amdgcn_s_wait_asynccnt(0);
#endif
}

// A-fragment (16x32 f16): row = lane&15, K = (lane>>4)*8 + {0..7} U {16..23}.
__device__ __forceinline__ v16h make_a(const _Float16* p) {
  v8h lo = *(const v8h*)p;
  v8h hi = *(const v8h*)(p + 16);
  v16h a;
#pragma unroll
  for (int e = 0; e < 8; ++e) { a[e] = lo[e]; a[e + 8] = hi[e]; }
  return a;
}
// B-fragment: K = (lane>>4)*8 + {0..7} U {16..23} (col = lane&15 folded into p by caller)
__device__ __forceinline__ v16h make_b16(const _Float16* p) {
  v8h lo = *(const v8h*)p;
  v8h hi = *(const v8h*)(p + 16);
  v16h b;
#pragma unroll
  for (int e = 0; e < 8; ++e) { b[e] = lo[e]; b[e + 8] = hi[e]; }
  return b;
}

// ---------------------------------------------------------------------------
// Generic WMMA GEMM:  C[M x (gridDim.y*64)] = epilogue(A[M x K] * Bt^T + bias)
// A: f16 row-major (lda).  Bt: f16 [N][K] (ldk).  B staged in LDS per 256-K
// slab (async), fragments read back as ds_load_b128 (padded stride = no-bank-
// conflict).  MODE bits: 1=ReLU, 2=BN-stats, 4=residual, 8=final f32 scatter.
// ---------------------------------------------------------------------------
#define GLDK 264  // 256 + 8 pad (halves)

template <int MODE>
__global__ __launch_bounds__(256) void gemm_ws(
    const _Float16* __restrict__ A, int lda,
    const _Float16* __restrict__ Bt, int ldk,
    const float* __restrict__ bias,
    _Float16* __restrict__ C, int ldc,
    const _Float16* __restrict__ Res,
    float* __restrict__ stats,
    float* __restrict__ outF,
    int M, int K) {
  constexpr bool RELU  = (MODE & 1) != 0;
  constexpr bool STATS = (MODE & 2) != 0;
  constexpr bool RESID = (MODE & 4) != 0;
  constexpr bool FINAL = (MODE & 8) != 0;

  __shared__ _Float16 lb[64 * GLDK];  // 33 KB

  const int lane = threadIdx.x & 31;
  const int wv   = threadIdx.x >> 5;
  const int r0   = (blockIdx.x * 8 + wv) * 16;
  const int n0   = blockIdx.y * 64;
  const int col  = lane & 15;
  const int kb   = (lane >> 4) * 8;
  const int kb2  = (lane >> 4) * 16;

  int arow = r0 + col;
  if (arow > M - 1) arow = M - 1;
  const _Float16* ap = A + (size_t)arow * lda + kb;

  v8f acc[4] = {};
  for (int ks = 0; ks < K; ks += 256) {
    const int kw  = (K - ks < 256) ? (K - ks) : 256;
    const int nch = kw >> 4;  // 16-half chunks per row
    for (int c = threadIdx.x; c < 64 * nch; c += 256) {
      const int n = c / nch, k16 = (c % nch) << 4;
      const _Float16* src = Bt + (size_t)(n0 + n) * ldk + ks + k16;
      _Float16* dst = &lb[n * GLDK + k16];
      cp16_async(src, dst);
      cp16_async(src + 8, dst + 8);
    }
    async_fence();
    __syncthreads();
    for (int kc = 0; kc < kw; kc += 32) {
      if (ks + kc + 32 < K) __builtin_prefetch(ap + ks + kc + 32, 0, 0);
      v16h a = make_a(ap + ks + kc);
      v16h bfr[4];  // batch all B ds_loads before the WMMA group (partial waits)
#pragma unroll
      for (int t = 0; t < 4; ++t)
        bfr[t] = make_b16(&lb[(t * 16 + col) * GLDK + kc + kb]);
#pragma unroll
      for (int t = 0; t < 4; ++t) acc[t] = WMMA_F16(a, bfr[t], acc[t]);
    }
    __syncthreads();
  }

  const int rbase = (lane >> 4) * 8;
#pragma unroll
  for (int t = 0; t < 4; ++t) {
    const int n = n0 + t * 16 + col;
    const float bi = bias ? bias[n] : 0.f;
    float s = 0.f, s2 = 0.f;
#pragma unroll
    for (int i = 0; i < 8; ++i) {
      const int r = r0 + rbase + i;
      float v = acc[t][i] + bi;
      if (r < M) {
        if (RESID) v += (float)Res[(size_t)r * ldc + n];
        if (RELU) v = fmaxf(v, 0.f);
        if (STATS) { s += v; s2 += v * v; }
        if (FINAL) {
          // r = h*200+w ; n = z*80+c ; out[c][h][w][z]
          const int h = r / 200, w = r % 200;
          const int z = n / 80, c = n % 80;
          outF[(((size_t)c * 200 + h) * 200 + w) * 8 + z] = v;
        } else {
          C[(size_t)r * ldc + n] = (_Float16)v;
        }
      }
    }
    if (STATS) {
      atomicAdd(&stats[2 * n], s);
      atomicAdd(&stats[2 * n + 1], s2);
    }
  }
}

// ---------------------------------------------------------------------------
// conv2: 3x3, 256->256 over 400x400, implicit GEMM on padded 402x402 image.
// 18 pipeline units (tap x K-half); weights double-buffered through LDS with
// async copies: stage unit u+1 while WMMAs consume unit u.  288 WMMAs/wave.
// ---------------------------------------------------------------------------
#define CLDK 136  // 128 + 8 pad (halves)

__global__ __launch_bounds__(256) void conv2_bn_stats(
    const _Float16* __restrict__ Xp,  // [402*402][256]
    const _Float16* __restrict__ Wt,  // [9][256 out][256 in]
    const float* __restrict__ bias,
    _Float16* __restrict__ Y,         // [160000][256]
    float* __restrict__ stats) {
  __shared__ _Float16 wbuf[2][64 * CLDK];  // 2 x 17 KB

  const int lane = threadIdx.x & 31;
  const int wv   = threadIdx.x >> 5;
  const int tile = blockIdx.x * 8 + wv;  // 10000 tiles: 400 rows x 25
  const int y    = tile / 25;
  const int x0   = (tile % 25) * 16;
  const int n0   = blockIdx.y * 64;
  const int col  = lane & 15;
  const int kb   = (lane >> 4) * 8;
  const int kb2  = (lane >> 4) * 16;

  auto stage = [&](int u) {
    const int tap = u >> 1, kh = (u & 1) << 7;
    for (int c = threadIdx.x; c < 64 * 8; c += 256) {
      const int n = c >> 3, k16 = (c & 7) << 4;
      const _Float16* src = Wt + ((size_t)tap * 256 + n0 + n) * 256 + kh + k16;
      _Float16* dst = &wbuf[u & 1][n * CLDK + k16];
      cp16_async(src, dst);
      cp16_async(src + 8, dst + 8);
    }
  };

  v8f acc[4] = {};
  stage(0);
  async_fence();
  __syncthreads();
  for (int u = 0; u < 18; ++u) {
    if (u < 17) stage(u + 1);  // overlap with compute below
    const int tap = u >> 1, kh = (u & 1) << 7;
    const int dy = tap / 3, dx = tap % 3;
    const _Float16* ap = Xp + ((size_t)(y + dy) * 402 + x0 + col + dx) * 256 + kh + kb;
    for (int kc = 0; kc < 128; kc += 32) {
      v16h a = make_a(ap + kc);
      v16h bfr[4];
#pragma unroll
      for (int t = 0; t < 4; ++t)
        bfr[t] = make_b16(&wbuf[u & 1][(t * 16 + col) * CLDK + kc + kb]);
#pragma unroll
      for (int t = 0; t < 4; ++t) acc[t] = WMMA_F16(a, bfr[t], acc[t]);
    }
    async_fence();
    __syncthreads();
  }

  const int rbase = (lane >> 4) * 8;
#pragma unroll
  for (int t = 0; t < 4; ++t) {
    const int n = n0 + t * 16 + col;
    const float bi = bias[n];
    float s = 0.f, s2 = 0.f;
#pragma unroll
    for (int i = 0; i < 8; ++i) {
      float v = acc[t][i] + bi;
      s += v; s2 += v * v;
      Y[((size_t)y * 400 + x0 + rbase + i) * 256 + n] = (_Float16)v;
    }
    atomicAdd(&stats[2 * n], s);
    atomicAdd(&stats[2 * n + 1], s2);
  }
}

// ---------------------------------------------------------------------------
// Window cross attention, flash-style.  One block = one (window, head):
// K staged token-major, V staged dim-major (so the P@V B-fragment is a
// pair of contiguous LDS v8h), 4 waves sweep the 25 query tiles.  V tail (tokens
// 400..415) zeroed so masked keys cannot inject NaN via WMMA 0*NaN.
// ---------------------------------------------------------------------------
#define ATT_KS 40   // 32 + 8 pad
#define ATT_VS 416  // 400 keys + 16 masked tail

__device__ __forceinline__ int pixw(int wy, int wx, int t) {
  return (wy * 20 + t / 20) * 200 + wx * 20 + (t % 20);
}

__global__ __launch_bounds__(128) void attn_kernel(
    const _Float16* __restrict__ Qp, const _Float16* __restrict__ Kp,
    const _Float16* __restrict__ Vp, _Float16* __restrict__ O) {
  __shared__ _Float16 Kt[400 * ATT_KS];   // 32 KB
  __shared__ _Float16 Vt[32 * ATT_VS];    // 26 KB
  __shared__ _Float16 pbuf[4][16][32];    // 4 KB
  const int tid  = threadIdx.x;
  const int lane = tid & 31;
  const int wv   = tid >> 5;              // 0..3
  const int head = blockIdx.x & 7;
  const int win  = blockIdx.x >> 3;
  const int wy = win / 10, wx = win % 10;
  const int hb = head * 32;
  const int col = lane & 15;
  const int kbA = (lane >> 4) * 8;
  const int kb2 = (lane >> 4) * 16;

  for (int i = tid; i < 32 * 16; i += 128)
    Vt[(i >> 4) * ATT_VS + 400 + (i & 15)] = (_Float16)0.f;
  for (int t = tid; t < 400; t += 128) {
    const size_t px = (size_t)pixw(wy, wx, t) * 256 + hb;
    v8h k0 = *(const v8h*)(Kp + px);
    v8h k1 = *(const v8h*)(Kp + px + 8);
    v8h k2 = *(const v8h*)(Kp + px + 16);
    v8h k3 = *(const v8h*)(Kp + px + 24);
    *(v8h*)&Kt[t * ATT_KS + 0]  = k0;
    *(v8h*)&Kt[t * ATT_KS + 8]  = k1;
    *(v8h*)&Kt[t * ATT_KS + 16] = k2;
    *(v8h*)&Kt[t * ATT_KS + 24] = k3;
    v8h v0 = *(const v8h*)(Vp + px);
    v8h v1 = *(const v8h*)(Vp + px + 8);
    v8h v2 = *(const v8h*)(Vp + px + 16);
    v8h v3 = *(const v8h*)(Vp + px + 24);
#pragma unroll
    for (int e = 0; e < 8; ++e) {
      Vt[(e)      * ATT_VS + t] = v0[e];
      Vt[(e + 8)  * ATT_VS + t] = v1[e];
      Vt[(e + 16) * ATT_VS + t] = v2[e];
      Vt[(e + 24) * ATT_VS + t] = v3[e];
    }
  }
  __syncthreads();

  for (int qt = wv; qt < 25; qt += 4) {
    const int t0 = qt * 16;
    v16h qa;  // Q fragment pre-scaled by 1/sqrt(HD)
    {
      const float sc = 0.17677669529663687f;
      const _Float16* qp = Qp + (size_t)pixw(wy, wx, t0 + col) * 256 + hb + kbA;
      v8h lo = *(const v8h*)qp;
      v8h hi = *(const v8h*)(qp + 16);
#pragma unroll
      for (int e = 0; e < 8; ++e) {
        qa[e]     = (_Float16)((float)lo[e] * sc);
        qa[e + 8] = (_Float16)((float)hi[e] * sc);
      }
    }
    float mrun[8], lrun[8];
#pragma unroll
    for (int i = 0; i < 8; ++i) { mrun[i] = -1e30f; lrun[i] = 0.f; }
    v8f oacc[2] = {};

    for (int kcv = 0; kcv < 13; ++kcv) {
      const int keybase = kcv * 32;
      v16h kfr[2];  // batch both K-fragment ds_loads before the WMMA pair
#pragma unroll
      for (int half = 0; half < 2; ++half) {
        const int key = keybase + half * 16 + col;
        const int tok = key < 400 ? key : 399;
        kfr[half] = make_b16(&Kt[tok * ATT_KS + kbA]);
      }
      v8f s[2] = {};
#pragma unroll
      for (int half = 0; half < 2; ++half) s[half] = WMMA_F16(qa, kfr[half], s[half]);
#pragma unroll
      for (int half = 0; half < 2; ++half)
        if (keybase + half * 16 + col >= 400) {
#pragma unroll
          for (int i = 0; i < 8; ++i) s[half][i] = -1e30f;
        }
      // online softmax per row (row = (lane>>4)*8 + i; 16 lanes per row)
#pragma unroll
      for (int i = 0; i < 8; ++i) {
        float mx = fmaxf(s[0][i], s[1][i]);
#pragma unroll
        for (int d = 1; d < 16; d <<= 1) mx = fmaxf(mx, __shfl_xor(mx, d, 32));
        const float mnew = fmaxf(mrun[i], mx);
        const float corr = __expf(mrun[i] - mnew);
        const float p0 = __expf(s[0][i] - mnew);
        const float p1 = __expf(s[1][i] - mnew);
        float rs = p0 + p1;
#pragma unroll
        for (int d = 1; d < 16; d <<= 1) rs += __shfl_xor(rs, d, 32);
        lrun[i] = lrun[i] * corr + rs;
        mrun[i] = mnew;
        oacc[0][i] *= corr;
        oacc[1][i] *= corr;
        const int row = (lane >> 4) * 8 + i;
        pbuf[wv][row][col]      = (_Float16)p0;
        pbuf[wv][row][col + 16] = (_Float16)p1;
      }
      // P(16x32) @ V(32x32); same-wave DS ops are in-order
      v16h vfr[2];
#pragma unroll
      for (int t2 = 0; t2 < 2; ++t2)
        vfr[t2] = make_b16(&Vt[(t2 * 16 + col) * ATT_VS + keybase + kbA]);
      v16h pa = make_a(&pbuf[wv][lane & 15][0] + kbA);
#pragma unroll
      for (int t2 = 0; t2 < 2; ++t2) oacc[t2] = WMMA_F16(pa, vfr[t2], oacc[t2]);
    }
#pragma unroll
    for (int t2 = 0; t2 < 2; ++t2)
#pragma unroll
      for (int i = 0; i < 8; ++i) {
        const int row = (lane >> 4) * 8 + i;
        const float ov = oacc[t2][i] / lrun[i];
        O[(size_t)pixw(wy, wx, t0 + row) * 256 + hb + t2 * 16 + col] = (_Float16)ov;
      }
  }
}

// --------------------------- elementwise helpers ---------------------------
__global__ void cvt_pad_k(const float* __restrict__ in, _Float16* __restrict__ out,
                          int M, int Kin, int Kpad) {
  int idx = blockIdx.x * blockDim.x + threadIdx.x;
  if (idx >= M * Kpad) return;
  int r = idx / Kpad, k = idx % Kpad;
  out[idx] = (k < Kin) ? (_Float16)in[(size_t)r * Kin + k] : (_Float16)0.f;
}

// in f32 [R][C] -> out f16 [C][Rpad] (N-major "Bt" weight layout, K zero-pad)
__global__ void cvt_tr_k(const float* __restrict__ in, _Float16* __restrict__ out,
                         int R, int C, int Rpad) {
  int idx = blockIdx.x * blockDim.x + threadIdx.x;
  if (idx >= C * Rpad) return;
  int c = idx / Rpad, r = idx % Rpad;
  out[idx] = (r < R) ? (_Float16)in[(size_t)r * C + c] : (_Float16)0.f;
}

// conv2_w (O,I,3,3) -> [tap][O][I] f16
__global__ void cvt_c2w_k(const float* __restrict__ in, _Float16* __restrict__ out) {
  int idx = blockIdx.x * blockDim.x + threadIdx.x;
  if (idx >= 9 * 256 * 256) return;
  int tap = idx >> 16, rem = idx & 65535;
  int o = rem >> 8, i = rem & 255;
  out[idx] = (_Float16)in[((size_t)o * 256 + i) * 9 + tap];
}

// bev (1,80,200,200,8) f32 -> [pixel][z*80+c] f16
__global__ void bev_t_k(const float* __restrict__ bev, _Float16* __restrict__ out) {
  int idx = blockIdx.x * blockDim.x + threadIdx.x;
  if (idx >= 40000 * 640) return;
  int p = idx / 640, j = idx % 640;
  int h = p / 200, w = p % 200;
  int z = j / 80, c = j % 80;
  out[idx] = (_Float16)bev[(((size_t)c * 200 + h) * 200 + w) * 8 + z];
}

// sparse-to-dense voxel scatter into [pixel][cv*8+z] f16
__global__ void scatter_k(const _Float16* __restrict__ f2, const int* __restrict__ coords,
                          _Float16* __restrict__ grid) {
  int idx = blockIdx.x * blockDim.x + threadIdx.x;
  if (idx >= 100000 * 64) return;
  int n = idx >> 6, cv = idx & 63;
  int cx = coords[3 * n], cy = coords[3 * n + 1], cz = coords[3 * n + 2];
  grid[((size_t)cy * 400 + cx) * 512 + cv * 8 + cz] = f2[idx];
}

// stats -> per-channel scale/shift
__global__ void bn_prep_k(const float* __restrict__ stats, const float* __restrict__ g,
                          const float* __restrict__ b, float* __restrict__ out, float inv_cnt) {
  int c = threadIdx.x;
  float mean = stats[2 * c] * inv_cnt;
  float var  = stats[2 * c + 1] * inv_cnt - mean * mean;
  float sc = g[c] * rsqrtf(var + 1e-5f);
  out[c] = sc;
  out[256 + c] = b[c] - mean * sc;
}

// bn1 + relu, write into zero-padded 402x402 image
__global__ void bn1_apply_k(const _Float16* __restrict__ y1, const float* __restrict__ bnp,
                            _Float16* __restrict__ xp) {
  int idx = blockIdx.x * blockDim.x + threadIdx.x;
  if (idx >= 160000 * 256) return;
  int p = idx >> 8, c = idx & 255;
  int y = p / 400, x = p % 400;
  float v = fmaxf((float)y1[idx] * bnp[c] + bnp[256 + c], 0.f);
  xp[((size_t)(y + 1) * 402 + (x + 1)) * 256 + c] = (_Float16)v;
}

// bn2 + relu + 2x2 maxpool -> prior_bev [40000][256] f16
__global__ void bn2_pool_k(const _Float16* __restrict__ y2, const float* __restrict__ bnp,
                           _Float16* __restrict__ pb) {
  int idx = blockIdx.x * blockDim.x + threadIdx.x;
  if (idx >= 40000 * 256) return;
  int p = idx >> 8, c = idx & 255;
  int oh = p / 200, ow = p % 200;
  float sc = bnp[c], sh = bnp[256 + c];
  float m = 0.f;  // relu(max) == max(0, max)
#pragma unroll
  for (int dy = 0; dy < 2; ++dy)
#pragma unroll
    for (int dx = 0; dx < 2; ++dx) {
      float v = (float)y2[((size_t)(2 * oh + dy) * 400 + 2 * ow + dx) * 256 + c] * sc + sh;
      m = fmaxf(m, v);
    }
  pb[idx] = (_Float16)m;
}

// ---------------------------------------------------------------------------
extern "C" void kernel_launch(void* const* d_in, const int* in_sizes, int n_in,
                              void* d_out, int out_size, void* d_ws, size_t ws_size,
                              hipStream_t stream) {
  (void)in_sizes; (void)n_in; (void)out_size; (void)ws_size;
  const float* bev    = (const float*)d_in[0];
  const float* prior  = (const float*)d_in[1];
  const int*   coords = (const int*)d_in[2];
  const float* w1 = (const float*)d_in[3];   const float* b1 = (const float*)d_in[4];
  const float* w2 = (const float*)d_in[5];   const float* b2 = (const float*)d_in[6];
  const float* c1wF = (const float*)d_in[7]; const float* c1b = (const float*)d_in[8];
  const float* g1 = (const float*)d_in[9];   const float* be1 = (const float*)d_in[10];
  const float* c2wF = (const float*)d_in[11]; const float* c2b = (const float*)d_in[12];
  const float* g2 = (const float*)d_in[13];  const float* be2 = (const float*)d_in[14];
  const float* wq = (const float*)d_in[15];  const float* bq = (const float*)d_in[16];
  const float* wk = (const float*)d_in[17];  const float* bk = (const float*)d_in[18];
  const float* wvw = (const float*)d_in[19]; const float* bv = (const float*)d_in[20];
  const float* wo = (const float*)d_in[21];  const float* bo = (const float*)d_in[22];
  const float* winw = (const float*)d_in[23]; const float* winb = (const float*)d_in[24];
  const float* woutw = (const float*)d_in[25]; const float* woutb = (const float*)d_in[26];
  float* out = (float*)d_out;

  // ---- workspace carve-out (f16 activations/weights, f32 stats) ----
  char* base = (char*)d_ws;
  size_t off = 0;
  auto take = [&](size_t bytes) -> char* {
    char* p = base + off;
    off += (bytes + 255) & ~(size_t)255;
    return p;
  };
  _Float16* fA    = (_Float16*)take((size_t)100000 * 96 * 2);
  _Float16* w1t   = (_Float16*)take((size_t)64 * 96 * 2);
  _Float16* f1    = (_Float16*)take((size_t)100000 * 64 * 2);
  _Float16* w2t   = (_Float16*)take((size_t)64 * 64 * 2);
  _Float16* f2    = (_Float16*)take((size_t)100000 * 64 * 2);
  _Float16* grid  = (_Float16*)take((size_t)160000 * 512 * 2);  // reused later as y2
  _Float16* c1w   = (_Float16*)take((size_t)256 * 512 * 2);
  _Float16* y1    = (_Float16*)take((size_t)160000 * 256 * 2);
  float*    stats1 = (float*)take(512 * 4);
  float*    bnp1   = (float*)take(512 * 4);
  _Float16* x2p   = (_Float16*)take((size_t)402 * 402 * 256 * 2);
  _Float16* c2w   = (_Float16*)take((size_t)9 * 256 * 256 * 2);
  float*    stats2 = (float*)take(512 * 4);
  float*    bnp2   = (float*)take(512 * 4);
  _Float16* pb    = (_Float16*)take((size_t)40000 * 256 * 2);
  _Float16* bevt  = (_Float16*)take((size_t)40000 * 640 * 2);
  _Float16* winT  = (_Float16*)take((size_t)256 * 640 * 2);
  _Float16* qbuf  = (_Float16*)take((size_t)40000 * 256 * 2);
  _Float16* wqT   = (_Float16*)take((size_t)256 * 256 * 2);
  _Float16* wkT   = (_Float16*)take((size_t)256 * 256 * 2);
  _Float16* wvT   = (_Float16*)take((size_t)256 * 256 * 2);
  _Float16* woT   = (_Float16*)take((size_t)256 * 256 * 2);
  _Float16* Qp    = (_Float16*)take((size_t)40000 * 256 * 2);
  _Float16* Kp    = (_Float16*)take((size_t)40000 * 256 * 2);
  _Float16* Vp    = (_Float16*)take((size_t)40000 * 256 * 2);
  _Float16* Obuf  = (_Float16*)take((size_t)40000 * 256 * 2);
  _Float16* resb  = (_Float16*)take((size_t)40000 * 256 * 2);
  _Float16* woutT = (_Float16*)take((size_t)640 * 256 * 2);
  _Float16* y2 = grid;  // conv1 input no longer needed when conv2 writes

  auto ew = [](long long n) { return dim3((unsigned)((n + 255) / 256)); };
  const dim3 blk(256);

  // ---- zero-init (graph-capturable) ----
  (void)hipMemsetAsync(grid, 0, (size_t)160000 * 512 * 2, stream);
  (void)hipMemsetAsync(x2p, 0, (size_t)402 * 402 * 256 * 2, stream);
  (void)hipMemsetAsync(stats1, 0, 512 * 4, stream);
  (void)hipMemsetAsync(stats2, 0, 512 * 4, stream);

  // ---- weight / input conversion to f16 ----
  cvt_pad_k<<<ew(100000LL * 96), blk, 0, stream>>>(prior, fA, 100000, 68, 96);
  cvt_tr_k<<<ew(64 * 96), blk, 0, stream>>>(w1, w1t, 68, 64, 96);
  cvt_tr_k<<<ew(64 * 64), blk, 0, stream>>>(w2, w2t, 64, 64, 64);
  cvt_pad_k<<<ew(256 * 512), blk, 0, stream>>>(c1wF, c1w, 256, 512, 512);  // (O,I) is already [N][K]
  cvt_c2w_k<<<ew(9 * 256 * 256), blk, 0, stream>>>(c2wF, c2w);
  cvt_tr_k<<<ew(256 * 640), blk, 0, stream>>>(winw, winT, 640, 256, 640);
  cvt_tr_k<<<ew(256 * 256), blk, 0, stream>>>(wq, wqT, 256, 256, 256);
  cvt_tr_k<<<ew(256 * 256), blk, 0, stream>>>(wk, wkT, 256, 256, 256);
  cvt_tr_k<<<ew(256 * 256), blk, 0, stream>>>(wvw, wvT, 256, 256, 256);
  cvt_tr_k<<<ew(256 * 256), blk, 0, stream>>>(wo, woT, 256, 256, 256);
  cvt_tr_k<<<ew(640 * 256), blk, 0, stream>>>(woutw, woutT, 256, 640, 256);
  bev_t_k<<<ew(40000LL * 640), blk, 0, stream>>>(bev, bevt);

  // ---- voxel MLP (relu(relu(x@w1+b1)@w2+b2)) ----
  gemm_ws<1><<<dim3(782, 1), blk, 0, stream>>>(fA, 96, w1t, 96, b1, f1, 64,
                                               nullptr, nullptr, nullptr, 100000, 96);
  gemm_ws<1><<<dim3(782, 1), blk, 0, stream>>>(f1, 64, w2t, 64, b2, f2, 64,
                                               nullptr, nullptr, nullptr, 100000, 64);
  scatter_k<<<ew(100000LL * 64), blk, 0, stream>>>(f2, coords, grid);

  // ---- conv1 (1x1, 512->256) + BN stats ----
  gemm_ws<2><<<dim3(1250, 4), blk, 0, stream>>>(grid, 512, c1w, 512, c1b, y1, 256,
                                                nullptr, stats1, nullptr, 160000, 512);
  bn_prep_k<<<1, 256, 0, stream>>>(stats1, g1, be1, bnp1, 1.f / 160000.f);
  bn1_apply_k<<<ew(160000LL * 256), blk, 0, stream>>>(y1, bnp1, x2p);

  // ---- conv2 (3x3, 256->256) + BN stats, then bn+relu+maxpool ----
  conv2_bn_stats<<<dim3(1250, 4), blk, 0, stream>>>(x2p, c2w, c2b, y2, stats2);
  bn_prep_k<<<1, 256, 0, stream>>>(stats2, g2, be2, bnp2, 1.f / 160000.f);
  bn2_pool_k<<<ew(40000LL * 256), blk, 0, stream>>>(y2, bnp2, pb);

  // ---- bev in-projection, QKV projections ----
  gemm_ws<0><<<dim3(313, 4), blk, 0, stream>>>(bevt, 640, winT, 640, winb, qbuf, 256,
                                               nullptr, nullptr, nullptr, 40000, 640);
  gemm_ws<0><<<dim3(313, 4), blk, 0, stream>>>(qbuf, 256, wqT, 256, bq, Qp, 256,
                                               nullptr, nullptr, nullptr, 40000, 256);
  gemm_ws<0><<<dim3(313, 4), blk, 0, stream>>>(pb, 256, wkT, 256, bk, Kp, 256,
                                               nullptr, nullptr, nullptr, 40000, 256);
  gemm_ws<0><<<dim3(313, 4), blk, 0, stream>>>(pb, 256, wvT, 256, bv, Vp, 256,
                                               nullptr, nullptr, nullptr, 40000, 256);

  // ---- window cross attention (800 blocks = 100 windows x 8 heads) ----
  attn_kernel<<<dim3(800), dim3(128), 0, stream>>>(Qp, Kp, Vp, Obuf);

  // ---- output projection + residual, final projection + scatter to NCHWZ ----
  gemm_ws<4><<<dim3(313, 4), blk, 0, stream>>>(Obuf, 256, woT, 256, bo, resb, 256,
                                               qbuf, nullptr, nullptr, 40000, 256);
  gemm_ws<8><<<dim3(313, 10), blk, 0, stream>>>(resb, 256, woutT, 256, woutb, nullptr, 0,
                                                nullptr, nullptr, out, 40000, 256);
}
